// static_heto_graph_23192823399229
// MI455X (gfx1250) — matrix operand order repaired and checked by reading, not compiled, not run on hardware
//
#include <hip/hip_runtime.h>
#include <stdint.h>
#include <stddef.h>

namespace {
constexpr int HDIM = 128;
constexpr int kNw = 40000, kNt = 400, kNd = 1600, kB = 8, kDPG = 200;
constexpr int kEww = 300000, kEwt = 150000, kEwd = 300000, kEtd = 20000, kEtt = 4000;
}

typedef __attribute__((ext_vector_type(16))) _Float16 v16h;
typedef __attribute__((ext_vector_type(8)))  float    v8f;
typedef int v4i_gcc __attribute__((vector_size(16)));   // matches builtin param type

// CDNA5 async global->LDS path (device pass only; plain-load fallback otherwise)
#if defined(__gfx1250__)
#if __has_builtin(__builtin_amdgcn_global_load_async_to_lds_b128) && \
    __has_builtin(__builtin_amdgcn_s_wait_asynccnt)
#define USE_ASYNC_LDS 1
#endif
#endif

// ---------------------------------------------------------------------------
// Y[N x 128] = X[N x 128] @ W[128 x 128] + b, via v_wmma_f32_16x16x32_f16.
// Wt is W transposed to [N=col][K] in f16.  One block = 16 rows, 8 waves,
// wave w computes the 16x16 tile at columns [16w,16w+16).
// ---------------------------------------------------------------------------
__global__ __launch_bounds__(256) void hgnn_gemm_tile16(
    const float* __restrict__ X, const _Float16* __restrict__ Wt,
    const float* __restrict__ bias, float* __restrict__ Y) {
  __shared__ alignas(16) float tile[16][HDIM + 4];  // +4 floats: bank-conflict pad
  const int rowBase = blockIdx.x * 16;
  const int tid = threadIdx.x;

#ifdef USE_ASYNC_LDS
  {
    int c = tid;  // 512 chunks of 16B (16 rows x 512B)
#pragma unroll
    for (int it = 0; it < 2; ++it, c += 256) {
      const int r = c >> 5;
      const int q = (c & 31) << 2;
      __builtin_amdgcn_global_load_async_to_lds_b128(
          (__attribute__((address_space(1))) v4i_gcc*)(uintptr_t)(
              X + (size_t)(rowBase + r) * HDIM + q),
          (__attribute__((address_space(3))) v4i_gcc*)(uintptr_t)(&tile[r][q]),
          0, 0);
    }
    __builtin_amdgcn_s_wait_asynccnt(0);
  }
  __syncthreads();
#else
  for (int i = tid; i < 16 * HDIM; i += 256) {
    const int r = i >> 7, q = i & (HDIM - 1);
    tile[r][q] = X[(size_t)(rowBase + r) * HDIM + q];
  }
  __syncthreads();
#endif

  const int lane = tid & 31;
  const int wave = tid >> 5;
  const int l16  = lane & 15;
  const int hi   = lane >> 4;      // which half-wave
  const int col  = wave * 16 + l16;

  v8f acc = {};
  const float* arow = &tile[l16][0];
  const _Float16* bcol = Wt + (size_t)col * HDIM;
#pragma unroll
  for (int k = 0; k < 4; ++k) {
    // A fragment: row=l16, K = ka..ka+15, ka = 32k + 16*hi
    const int ka = 32 * k + 16 * hi;
    v16h a, b;
#pragma unroll
    for (int i = 0; i < 8; ++i) {
      a[i]     = (_Float16)arow[ka + i];
      a[i + 8] = (_Float16)arow[ka + 8 + i];
    }
    // B fragment: col=l16, K = 32k + 16*hi .. +15 (contiguous in Wt)
    b = *(const v16h*)(bcol + 32 * k + 16 * hi);
    acc = __builtin_amdgcn_wmma_f32_16x16x32_f16(false, a, false, b,
                                                 (short)0, acc, false, false);
  }
  const float bb = bias[col];
#pragma unroll
  for (int v = 0; v < 8; ++v) {  // D: M = v + 8*hi, N = col
    Y[(size_t)(rowBase + v + 8 * hi) * HDIM + col] = acc[v] + bb;
  }
}

// Wt[n][k] = (f16) W[k][n]
__global__ void hgnn_cvt_wt(const float* __restrict__ W, _Float16* __restrict__ Wt) {
  const int n = blockIdx.x, k = threadIdx.x;
  Wt[(size_t)n * HDIM + k] = (_Float16)W[(size_t)k * HDIM + n];
}

__global__ void hgnn_zero(float* __restrict__ p, int n) {
  const int i = blockIdx.x * 256 + threadIdx.x;
  if (i < n) p[i] = 0.0f;
}

__global__ void hgnn_gather_rows(const float* __restrict__ tab, const int* __restrict__ ids,
                                 float* __restrict__ out, int n) {
  const int i = blockIdx.x * 256 + threadIdx.x;
  if (i < n * HDIM) out[i] = tab[(size_t)ids[i >> 7] * HDIM + (i & (HDIM - 1))];
}

__global__ void hgnn_count_deg(const int* __restrict__ dst, float* __restrict__ deg, int E) {
  const int e = blockIdx.x * 256 + threadIdx.x;
  if (e < E) atomicAdd(&deg[dst[e]], 1.0f);
}

// acc[dst[e]] += feat[src[e]] * w[e]; 2 edges per 256-thread block
__global__ void hgnn_scatter(const float* __restrict__ feat, const int* __restrict__ src,
                             const int* __restrict__ dst, const float* __restrict__ ew,
                             float* __restrict__ acc, int E) {
  const int e = blockIdx.x * 2 + (threadIdx.x >> 7);
  const int t = threadIdx.x & (HDIM - 1);
  if (e >= E) return;
  const int s = src[e], d = dst[e];
  const float w = ew[e];
  atomicAdd(&acc[(size_t)d * HDIM + t], feat[(size_t)s * HDIM + t] * w);
}

__global__ void hgnn_mean_div(float* __restrict__ acc, const float* __restrict__ deg, int n) {
  const int i = blockIdx.x * 256 + threadIdx.x;
  if (i < n * HDIM) {
    const float d = deg[i >> 7];
    acc[i] = d > 0.0f ? acc[i] / d : 0.0f;
  }
}

__global__ void hgnn_mean2_add(const float* __restrict__ a1, const float* __restrict__ d1,
                               const float* __restrict__ a2, const float* __restrict__ d2,
                               float* __restrict__ out, int n) {
  const int i = blockIdx.x * 256 + threadIdx.x;
  if (i < n * HDIM) {
    const float c1 = d1[i >> 7], c2 = d2[i >> 7];
    const float x1 = c1 > 0.0f ? a1[i] / c1 : 0.0f;
    const float x2 = c2 > 0.0f ? a2[i] / c2 : 0.0f;
    out[i] = x1 + x2;
  }
}

// doc pool -> logits -> BCE loss + sigmoid.  1 block, 8 waves, wave b = graph b.
__global__ __launch_bounds__(256) void hgnn_head(
    const float* __restrict__ hd, const float* __restrict__ oW,
    const float* __restrict__ ob, const float* __restrict__ y,
    float* __restrict__ out) {
  __shared__ float zsh[kB];
  const int wave = threadIdx.x >> 5, lane = threadIdx.x & 31;
  const float w0 = oW[lane], w1 = oW[lane + 32], w2 = oW[lane + 64], w3 = oW[lane + 96];
  float acc = 0.0f;
  const float* base = hd + (size_t)wave * kDPG * HDIM;
  for (int d = 0; d < kDPG; ++d) {
    const float* r = base + (size_t)d * HDIM;
    acc += r[lane] * w0 + r[lane + 32] * w1 + r[lane + 64] * w2 + r[lane + 96] * w3;
  }
  for (int off = 16; off > 0; off >>= 1) acc += __shfl_down(acc, off, 32);
  if (lane == 0) zsh[wave] = acc / (float)kDPG + ob[0];
  __syncthreads();
  if (threadIdx.x == 0) {
    float loss = 0.0f;
    for (int b = 0; b < kB; ++b) {
      const float z = zsh[b];
      loss += fmaxf(z, 0.0f) - z * y[b] + log1pf(expf(-fabsf(z)));
      out[1 + b] = 1.0f / (1.0f + expf(-z));
    }
    out[0] = loss / (float)kB;
  }
}

// ---------------------------------------------------------------------------
extern "C" void kernel_launch(void* const* d_in, const int* in_sizes, int n_in,
                              void* d_out, int out_size, void* d_ws, size_t ws_size,
                              hipStream_t stream) {
  (void)in_sizes; (void)n_in; (void)out_size; (void)ws_size;

  const int*   word_ids     = (const int*)d_in[0];
  const int*   topic_ids    = (const int*)d_in[1];
  const int*   ww_src = (const int*)d_in[2];  const int* ww_dst = (const int*)d_in[3];
  const float* ww_w   = (const float*)d_in[4];
  const int*   wt_src = (const int*)d_in[5];  const int* wt_dst = (const int*)d_in[6];
  const float* wt_w   = (const float*)d_in[7];
  const int*   wd_src = (const int*)d_in[8];  const int* wd_dst = (const int*)d_in[9];
  const float* wd_w   = (const float*)d_in[10];
  const int*   td_src = (const int*)d_in[11]; const int* td_dst = (const int*)d_in[12];
  const float* td_w   = (const float*)d_in[13];
  const int*   tt_src = (const int*)d_in[14]; const int* tt_dst = (const int*)d_in[15];
  const float* tt_w   = (const float*)d_in[16];
  const float* y_data = (const float*)d_in[17];
  const float* word_embeds  = (const float*)d_in[18];
  const float* topic_embeds = (const float*)d_in[19];
  // params{1,2}: setup_inputs() insertion order: ww, wt, wd, td, tt; W then b
  enum { SL_WW = 0, SL_WT = 1, SL_WD = 2, SL_TD = 3, SL_TT = 4 };
  auto Wp = [&](int set, int slot) { return (const float*)d_in[20 + set * 10 + slot * 2]; };
  auto Bp = [&](int set, int slot) { return (const float*)d_in[20 + set * 10 + slot * 2 + 1]; };
  const float* out_W = (const float*)d_in[40];
  const float* out_b = (const float*)d_in[41];
  float* out = (float*)d_out;

  // -------- workspace layout (256B-aligned slabs) --------
  char* base = (char*)d_ws;
  size_t off = 0;
  auto alloc = [&](size_t bytes) -> void* {
    void* p = base + off;
    off += (bytes + 255) & ~(size_t)255;
    return p;
  };
  float* HW    = (float*)alloc((size_t)kNw * HDIM * 4);
  float* TMPW  = (float*)alloc((size_t)kNw * HDIM * 4);
  float* HT    = (float*)alloc((size_t)kNt * HDIM * 4);
  float* TMPT  = (float*)alloc((size_t)kNt * HDIM * 4);
  float* HT2   = (float*)alloc((size_t)kNt * HDIM * 4);
  float* ACCT1 = (float*)alloc((size_t)kNt * HDIM * 4);
  float* ACCT2 = (float*)alloc((size_t)kNt * HDIM * 4);
  float* ACCD1 = (float*)alloc((size_t)kNd * HDIM * 4);
  float* ACCD2 = (float*)alloc((size_t)kNd * HDIM * 4);
  float* HDOC  = (float*)alloc((size_t)kNd * HDIM * 4);
  float* DEG   = (float*)alloc((size_t)(kNw + 2 * kNt + 2 * kNd) * 4);
  _Float16* WT = (_Float16*)alloc((size_t)10 * HDIM * HDIM * 2);
  float* deg_ww = DEG;
  float* deg_wt = DEG + kNw;
  float* deg_tt = DEG + kNw + kNt;
  float* deg_wd = DEG + kNw + 2 * kNt;
  float* deg_td = DEG + kNw + 2 * kNt + kNd;

  auto zero = [&](float* p, int n) {
    hgnn_zero<<<dim3((n + 255) / 256), dim3(256), 0, stream>>>(p, n);
  };
  auto gemm = [&](const float* X, int wtIdx, const float* bias, float* Y, int N) {
    hgnn_gemm_tile16<<<dim3(N / 16), dim3(256), 0, stream>>>(
        X, WT + (size_t)wtIdx * HDIM * HDIM, bias, Y);
  };
  auto scatter = [&](const float* feat, const int* src, const int* dst,
                     const float* ew, float* acc, int E) {
    hgnn_scatter<<<dim3((E + 1) / 2), dim3(256), 0, stream>>>(feat, src, dst, ew, acc, E);
  };
  auto count = [&](const int* dst, float* deg, int E) {
    hgnn_count_deg<<<dim3((E + 255) / 256), dim3(256), 0, stream>>>(dst, deg, E);
  };

  // -------- setup: weights->f16T, degrees, embedding gathers --------
  for (int s = 0; s < 2; ++s)
    for (int sl = 0; sl < 5; ++sl)
      hgnn_cvt_wt<<<dim3(HDIM), dim3(HDIM), 0, stream>>>(Wp(s, sl),
                                                         WT + (size_t)(s * 5 + sl) * HDIM * HDIM);
  zero(DEG, kNw + 2 * kNt + 2 * kNd);
  count(ww_dst, deg_ww, kEww);
  count(wt_dst, deg_wt, kEwt);
  count(tt_dst, deg_tt, kEtt);
  count(wd_dst, deg_wd, kEwd);
  count(td_dst, deg_td, kEtd);
  hgnn_gather_rows<<<dim3((kNw * HDIM + 255) / 256), dim3(256), 0, stream>>>(
      word_embeds, word_ids, HW, kNw);
  hgnn_gather_rows<<<dim3((kNt * HDIM + 255) / 256), dim3(256), 0, stream>>>(
      topic_embeds, topic_ids, HT, kNt);

  // -------- two GNN layers --------
  for (int s = 0; s < 2; ++s) {
    const bool last = (s == 1);
    // phase 1: wh = hw @ Www + b ; hw_new = segmean_ww(wh * w)
    gemm(HW, s * 5 + SL_WW, Bp(s, SL_WW), TMPW, kNw);
    zero(HW, kNw * HDIM);
    scatter(TMPW, ww_src, ww_dst, ww_w, HW, kEww);
    hgnn_mean_div<<<dim3((kNw * HDIM + 255) / 256), dim3(256), 0, stream>>>(HW, deg_ww, kNw);
    // phase 2 transforms: hw2 = Wwd(Wwt(hw_new)), ht2 = Wtt(Wtd(ht))
    gemm(HW, s * 5 + SL_WT, Bp(s, SL_WT), TMPW, kNw);
    gemm(TMPW, s * 5 + SL_WD, Bp(s, SL_WD), HW, kNw);   // HW = hw2
    gemm(HT, s * 5 + SL_TD, Bp(s, SL_TD), TMPT, kNt);
    gemm(TMPT, s * 5 + SL_TT, Bp(s, SL_TT), HT2, kNt);  // HT2 = ht2
    // ht_new = segmean_wt(hw2) + segmean_tt(ht2)
    zero(ACCT1, kNt * HDIM);
    zero(ACCT2, kNt * HDIM);
    scatter(HW, wt_src, wt_dst, wt_w, ACCT1, kEwt);
    scatter(HT2, tt_src, tt_dst, tt_w, ACCT2, kEtt);
    hgnn_mean2_add<<<dim3((kNt * HDIM + 255) / 256), dim3(256), 0, stream>>>(
        ACCT1, deg_wt, ACCT2, deg_tt, HT, kNt);         // HT = ht_new
    // hd_new needed only from the final layer
    if (last) {
      zero(ACCD1, kNd * HDIM);
      zero(ACCD2, kNd * HDIM);
      scatter(HW, wd_src, wd_dst, wd_w, ACCD1, kEwd);
      scatter(HT2, td_src, td_dst, td_w, ACCD2, kEtd);
      hgnn_mean2_add<<<dim3((kNd * HDIM + 255) / 256), dim3(256), 0, stream>>>(
          ACCD1, deg_wd, ACCD2, deg_td, HDOC, kNd);
    }
  }

  // -------- head: pool docs, logits, BCE loss, sigmoid --------
  hgnn_head<<<dim3(1), dim3(256), 0, stream>>>(HDOC, out_W, out_b, y_data, out);
}
